// UniPhyModel_22093311770922
// MI455X (gfx1250) — compile-verified, matrix operand order checked by reading, not run on hardware
//
#include <hip/hip_runtime.h>
#include <hip/hip_bf16.h>
#include <math.h>

typedef __attribute__((ext_vector_type(16))) _Float16 v16h;
typedef __attribute__((ext_vector_type(8)))  float    v8f;

#define NIMG  48        // B*T
#define BB    4
#define TT    12
#define DD    384
#define D2    768
#define DFF2  3072
#define MTOK  12288     // NIMG*16*16 tokens
#define KENC  1024
#define NDEC  1024

// ---------------------------------------------------------------------------
// WMMA fragment load: per ISA 16-bit A-matrix 16x32 layout, each lane owns two
// contiguous 8-half (16B) runs: [k0+ksel*8 .. +7] and [k0+16+ksel*8 .. +7].
// ---------------------------------------------------------------------------
__device__ __forceinline__ v16h load_frag(const _Float16* __restrict__ p) {
  v16h f;
#pragma unroll
  for (int j = 0; j < 8; ++j) f[j] = p[j];
#pragma unroll
  for (int j = 0; j < 8; ++j) f[8 + j] = p[16 + j];
  return f;
}

// C[M][N] = A[M][K] * BT[N][K]^T (+bias).
// Register-blocked: one 64x32 output block per wave (4x2 WMMA tiles,
// 8 v_wmma per K-step on 6 fragment loads -> 2.67x operand reuse).
__global__ void gemm64x32(const _Float16* __restrict__ A, const _Float16* __restrict__ BT,
                          const float* __restrict__ bias, float* __restrict__ C,
                          int M, int N, int K) {
  int wave = blockIdx.x * (blockDim.x >> 5) + (threadIdx.x >> 5);
  int tiles_n = N >> 5;
  if (wave >= (M >> 6) * tiles_n) return;     // wave-uniform: EXEC stays all-ones
  int tm = wave / tiles_n, tn = wave - tm * tiles_n;
  int lane = threadIdx.x & 31;
  int r16 = lane & 15, ksel = (lane >> 4) & 1;
  const _Float16* a0 = A  + (size_t)(tm * 64 + r16) * K + ksel * 8;
  const _Float16* b0 = BT + (size_t)(tn * 32 + r16) * K + ksel * 8;
  const size_t step16 = (size_t)16 * K;
  v8f acc[4][2] = {};
  for (int k0 = 0; k0 < K; k0 += 32) {
    v16h av[4], bv[2];
#pragma unroll
    for (int s = 0; s < 4; ++s) av[s] = load_frag(a0 + (size_t)s * step16 + k0);
#pragma unroll
    for (int t = 0; t < 2; ++t) bv[t] = load_frag(b0 + (size_t)t * step16 + k0);
#pragma unroll
    for (int s = 0; s < 4; ++s)
#pragma unroll
      for (int t = 0; t < 2; ++t)
        acc[s][t] = __builtin_amdgcn_wmma_f32_16x16x32_f16(false, av[s], false, bv[t],
                                                           (short)0, acc[s][t], false, false);
  }
#pragma unroll
  for (int t = 0; t < 2; ++t) {
    float bn = bias ? bias[tn * 32 + t * 16 + r16] : 0.0f;
#pragma unroll
    for (int s = 0; s < 4; ++s) {
      float* crow = C + (size_t)(tm * 64 + s * 16 + 8 * ksel) * N + tn * 32 + t * 16 + r16;
#pragma unroll
      for (int r = 0; r < 8; ++r) crow[(size_t)r * N] = acc[s][t][r] + bn;
    }
  }
}

// 3x3 SAME conv 768->768 as implicit GEMM over padded NHWC f16 input.
// Same 64x32 register blocking; the 4 M-subtiles are 4 consecutive token rows
// (y..y+3) of one image. 9 taps x 24 K-steps x 8 WMMA per wave.
__global__ void conv3x3_gemm(const _Float16* __restrict__ xnp, const _Float16* __restrict__ BT,
                             const float* __restrict__ bias, float* __restrict__ C) {
  int wave = blockIdx.x * (blockDim.x >> 5) + (threadIdx.x >> 5);
  const int tiles_n = D2 / 32;                    // 24
  if (wave >= (MTOK / 64) * tiles_n) return;
  int tm = wave / tiles_n, tn = wave - tm * tiles_n;
  int img = tm >> 2, ybase = (tm & 3) * 4;
  int lane = threadIdx.x & 31;
  int r16 = lane & 15, ksel = (lane >> 4) & 1;    // r16 = x coordinate / out-chan
  const size_t bstep16 = (size_t)16 * D2;
  v8f acc[4][2] = {};
#pragma unroll
  for (int dy = 0; dy < 3; ++dy)
    for (int dx = 0; dx < 3; ++dx) {
      const _Float16* arow0 = xnp + (((size_t)img * 18 + ybase + dy) * 18 + (r16 + dx)) * D2
                                  + ksel * 8;
      const _Float16* brow0 = BT + ((size_t)(dy * 3 + dx) * D2 + tn * 32 + r16) * D2
                                 + ksel * 8;
      for (int k0 = 0; k0 < D2; k0 += 32) {
        v16h av[4], bv[2];
#pragma unroll
        for (int s = 0; s < 4; ++s) av[s] = load_frag(arow0 + (size_t)s * 18 * D2 + k0);
#pragma unroll
        for (int t = 0; t < 2; ++t) bv[t] = load_frag(brow0 + (size_t)t * bstep16 + k0);
#pragma unroll
        for (int s = 0; s < 4; ++s)
#pragma unroll
          for (int t = 0; t < 2; ++t)
            acc[s][t] = __builtin_amdgcn_wmma_f32_16x16x32_f16(false, av[s], false, bv[t],
                                                               (short)0, acc[s][t],
                                                               false, false);
      }
    }
#pragma unroll
  for (int t = 0; t < 2; ++t) {
    float bn = bias[tn * 32 + t * 16 + r16];
#pragma unroll
    for (int s = 0; s < 4; ++s) {
      float* crow = C + (size_t)(tm * 64 + s * 16 + 8 * ksel) * D2 + tn * 32 + t * 16 + r16;
#pragma unroll
      for (int r = 0; r < 8; ++r) crow[(size_t)r * D2] = acc[s][t][r] + bn;
    }
  }
}

// ---------------------------------------------------------------------------
// Pointwise / layout kernels
// ---------------------------------------------------------------------------
__global__ void fill_h16(_Float16* p, int n) {
  int i = blockIdx.x * blockDim.x + threadIdx.x;
  if (i < n) p[i] = (_Float16)0.0f;
}
__global__ void cast_f32_h16(const float* __restrict__ s, _Float16* __restrict__ d, int n) {
  int i = blockIdx.x * blockDim.x + threadIdx.x;
  if (i < n) d[i] = (_Float16)s[i];
}
__global__ void add_inplace(float* __restrict__ d, const float* __restrict__ s, int n) {
  int i = blockIdx.x * blockDim.x + threadIdx.x;
  if (i < n) d[i] += s[i];
}
__global__ void gelu_h16(const float* __restrict__ s, _Float16* __restrict__ d, int n) {
  int i = blockIdx.x * blockDim.x + threadIdx.x;
  if (i < n) {
    float x = s[i];
    float g = 0.5f * x * (1.0f + tanhf(0.7978845608028654f * (x + 0.044715f * x * x * x)));
    d[i] = (_Float16)g;
  }
}

// unfold 16x16 patches: A[m=(n,gy,gx)][k=(cin,py,px)]
__global__ void unfold_enc(const float* __restrict__ x, _Float16* __restrict__ A) {
  long long id = (long long)blockIdx.x * blockDim.x + threadIdx.x;
  if (id >= (long long)MTOK * KENC) return;
  int k = (int)(id & 1023);
  int m = (int)(id >> 10);
  int n = m >> 8, gy = (m >> 4) & 15, gx = m & 15;
  int cin = k >> 8, py = (k >> 4) & 15, px = k & 15;
  A[id] = (_Float16)x[(((size_t)n * 4 + cin) * 256 + gy * 16 + py) * 256 + gx * 16 + px];
}

// Complex block weight, transposed: BT[2Nd][2Kd], BT[n][k] = W'[k][n],
// W' = [[Wr, Wi], [-Wi, Wr]] with Wr,Wi row-major [Kd][Nd].
__global__ void build_cplx_BT(const float* __restrict__ Wr, const float* __restrict__ Wi,
                              _Float16* __restrict__ BT, int Kd, int Nd) {
  long long id = (long long)blockIdx.x * blockDim.x + threadIdx.x;
  long long tot = (long long)4 * Kd * Nd;
  if (id >= tot) return;
  int K2 = 2 * Kd;
  int k = (int)(id % K2);
  int n = (int)(id / K2);
  float v;
  if (k < Kd) v = (n < Nd) ? Wr[(size_t)k * Nd + n] : Wi[(size_t)k * Nd + (n - Nd)];
  else        v = (n < Nd) ? -Wi[(size_t)(k - Kd) * Nd + n]
                           : Wr[(size_t)(k - Kd) * Nd + (n - Nd)];
  BT[id] = (_Float16)v;
}

// conv weights (o,c,3,3) -> BT[tap][o][c]
__global__ void build_conv_BT(const float* __restrict__ cw, _Float16* __restrict__ BT) {
  long long id = (long long)blockIdx.x * blockDim.x + threadIdx.x;
  if (id >= (long long)9 * D2 * D2) return;
  int c = (int)(id % D2);
  int o = (int)((id / D2) % D2);
  int tap = (int)(id / ((long long)D2 * D2));
  BT[id] = (_Float16)cw[((size_t)o * D2 + c) * 9 + tap];
}

// LayerNorm over 768 (wave per token) -> padded NHWC f16 (interior of 18x18)
__global__ void ln_pad(const float* __restrict__ z, const float* __restrict__ g,
                       const float* __restrict__ b, _Float16* __restrict__ xnp) {
  int token = blockIdx.x * (blockDim.x >> 5) + (threadIdx.x >> 5);
  if (token >= MTOK) return;
  int lane = threadIdx.x & 31;
  const float* row = z + (size_t)token * D2;
  float s = 0.f, s2 = 0.f;
  for (int c = lane; c < D2; c += 32) { float v = row[c]; s += v; s2 += v * v; }
  for (int off = 16; off; off >>= 1) { s += __shfl_xor(s, off, 32); s2 += __shfl_xor(s2, off, 32); }
  float mean = s * (1.f / D2);
  float rstd = rsqrtf(s2 * (1.f / D2) - mean * mean + 1e-5f);
  int img = token >> 8, y = (token >> 4) & 15, x = token & 15;
  _Float16* out = xnp + (((size_t)img * 18 + y + 1) * 18 + x + 1) * D2;
  for (int c = lane; c < D2; c += 32)
    out[c] = (_Float16)((row[c] - mean) * rstd * g[c] + b[c]);
}

// LayerNorm over 768 -> dense f16 rows
__global__ void ln_rows(const float* __restrict__ z, const float* __restrict__ g,
                        const float* __restrict__ b, _Float16* __restrict__ out) {
  int token = blockIdx.x * (blockDim.x >> 5) + (threadIdx.x >> 5);
  if (token >= MTOK) return;
  int lane = threadIdx.x & 31;
  const float* row = z + (size_t)token * D2;
  float s = 0.f, s2 = 0.f;
  for (int c = lane; c < D2; c += 32) { float v = row[c]; s += v; s2 += v * v; }
  for (int off = 16; off; off >>= 1) { s += __shfl_xor(s, off, 32); s2 += __shfl_xor(s2, off, 32); }
  float mean = s * (1.f / D2);
  float rstd = rsqrtf(s2 * (1.f / D2) - mean * mean + 1e-5f);
  _Float16* o = out + (size_t)token * D2;
  for (int c = lane; c < D2; c += 32)
    o[c] = (_Float16)((row[c] - mean) * rstd * g[c] + b[c]);
}

// x_mean[bt][c] = mean over 256 spatial tokens of x_eig[(bt*256+hw)][c]
__global__ void mean_hw(const float* __restrict__ xe, float* __restrict__ xm) {
  int i = blockIdx.x * blockDim.x + threadIdx.x;
  if (i >= NIMG * D2) return;
  int bt = i / D2, c = i - bt * D2;
  const float* p = xe + (size_t)bt * 256 * D2 + c;
  float s = 0.f;
  for (int hw = 0; hw < 256; ++hw) s += p[(size_t)hw * D2];
  xm[i] = s * (1.f / 256.f);
}

// flux scan over T (per (b,d)); flux stored [bt][re(384)||im(384)]
__global__ void flux_scan(const float* __restrict__ xm, const float* __restrict__ dtp,
                          const float* __restrict__ lamf, float* __restrict__ flux) {
  int i = blockIdx.x * blockDim.x + threadIdx.x;
  if (i >= BB * DD) return;
  int b = i / DD, d = i - b * DD;
  float lf = lamf[d];
  float sp = (lf > 20.f) ? lf : log1pf(expf(lf));
  float fr = 0.f, fi = 0.f;
  for (int t = 0; t < TT; ++t) {
    int bt = b * TT + t;
    float dt = dtp[bt];
    float a = expf(-sp * dt);
    fr = fr * a + xm[(size_t)bt * D2 + d] * dt;
    fi = fi * a + xm[(size_t)bt * D2 + DD + d] * dt;
    flux[(size_t)bt * D2 + d] = fr;
    flux[(size_t)bt * D2 + DD + d] = fi;
  }
}

// per (bt,e): source = flux @ Ws (complex matvec), gate, op_decay, op_forcing
// small layout: [0]=src_re [1]=src_im [2]=gate [3]=od_re [4]=od_im [5]=of_re [6]=of_im
__global__ void source_gate(const float* __restrict__ flux,
                            const float* __restrict__ Wsr, const float* __restrict__ Wsi,
                            const float* __restrict__ Wg, const float* __restrict__ nu,
                            const float* __restrict__ omega, const float* __restrict__ dtp,
                            float* __restrict__ sm) {
  int i = blockIdx.x * blockDim.x + threadIdx.x;
  const int NE = NIMG * DD;
  if (i >= NE) return;
  int bt = i / DD, e = i - bt * DD;
  const float* fr = flux + (size_t)bt * D2;
  const float* fi = fr + DD;
  float sr = 0.f, si = 0.f;
  for (int d = 0; d < DD; ++d) {
    float wr = Wsr[(size_t)d * DD + e], wi = Wsi[(size_t)d * DD + e];
    sr += fr[d] * wr - fi[d] * wi;
    si += fr[d] * wi + fi[d] * wr;
  }
  float gate = 1.f / (1.f + expf(-fr[e] * Wg[e]));
  float nv = nu[e];
  float spn = (nv > 20.f) ? nv : log1pf(expf(nv));
  float lre = fminf(fmaxf(-spn, -5.f), 0.3f);
  float lim = omega[e];
  float dt = dtp[bt];
  float er = expf(lre * dt);
  float odr = er * cosf(lim * dt), odi = er * sinf(lim * dt);
  float nr = odr - 1.f, ni = odi;
  float den = lre * lre + lim * lim;
  float ofr = (nr * lre + ni * lim) / den;
  float ofi = (ni * lre - nr * lim) / den;
  sm[0 * NE + i] = sr;   sm[1 * NE + i] = si;  sm[2 * NE + i] = gate;
  sm[3 * NE + i] = odr;  sm[4 * NE + i] = odi;
  sm[5 * NE + i] = ofr;  sm[6 * NE + i] = ofi;
}

// in-place: x_eig -> u_t = (x_eig*g + src*(1-g)) * op_forcing
__global__ void forcing_ut(float* __restrict__ xe, const float* __restrict__ sm) {
  long long i = (long long)blockIdx.x * blockDim.x + threadIdx.x;
  if (i >= (long long)MTOK * DD) return;
  int d = (int)(i % DD);
  int m = (int)(i / DD);
  int bt = m >> 8;
  const int NE = NIMG * DD;
  int q = bt * DD + d;
  float g = sm[2 * NE + q];
  float xr = xe[(size_t)m * D2 + d], xi = xe[(size_t)m * D2 + DD + d];
  float fr = xr * g + sm[0 * NE + q] * (1.f - g);
  float fi = xi * g + sm[1 * NE + q] * (1.f - g);
  float ofr = sm[5 * NE + q], ofi = sm[6 * NE + q];
  xe[(size_t)m * D2 + d]      = fr * ofr - fi * ofi;
  xe[(size_t)m * D2 + DD + d] = fr * ofi + fi * ofr;
}

// in-place scan over t per (b,hw,d): u = u*od + u_t
__global__ void uout_scan(float* __restrict__ ut, const float* __restrict__ sm) {
  int i = blockIdx.x * blockDim.x + threadIdx.x;
  if (i >= BB * 256 * DD) return;
  int b = i / (256 * DD);
  int rem = i - b * 256 * DD;
  int hw = rem / DD, d = rem - hw * DD;
  const int NE = NIMG * DD;
  float ur = 0.f, ui = 0.f;
  for (int t = 0; t < TT; ++t) {
    int bt = b * TT + t;
    size_t m = (size_t)bt * 256 + hw;
    float odr = sm[3 * NE + bt * DD + d], odi = sm[4 * NE + bt * DD + d];
    float tr = ut[m * D2 + d], ti = ut[m * D2 + DD + d];
    float nr = ur * odr - ui * odi + tr;
    float ni = ur * odi + ui * odr + ti;
    ur = nr; ui = ni;
    ut[m * D2 + d] = ur; ut[m * D2 + DD + d] = ui;
  }
}

// z += df + delta
__global__ void residual2(float* __restrict__ z, const float* __restrict__ a,
                          const float* __restrict__ b, int n) {
  int i = blockIdx.x * blockDim.x + threadIdx.x;
  if (i < n) z[i] += a[i] + b[i];
}

// y[bt][co][Y][X] = g2[(bt,gy,gx)][co*256+py*16+px]
__global__ void pixel_shuffle(const float* __restrict__ g2, float* __restrict__ out) {
  long long i = (long long)blockIdx.x * blockDim.x + threadIdx.x;
  if (i >= (long long)NIMG * 4 * 256 * 256) return;
  long long o = i;
  int X = (int)(o & 255); o >>= 8;
  int Y = (int)(o & 255); o >>= 8;
  int co = (int)(o & 3);  o >>= 2;
  int bt = (int)o;
  int gy = Y >> 4, py = Y & 15, gx = X >> 4, px = X & 15;
  out[i] = g2[((size_t)bt * 256 + gy * 16 + gx) * NDEC + co * 256 + py * 16 + px];
}

// ---------------------------------------------------------------------------
extern "C" void kernel_launch(void* const* d_in, const int* in_sizes, int n_in,
                              void* d_out, int out_size, void* d_ws, size_t ws_size,
                              hipStream_t stream) {
  const float* x      = (const float*)d_in[0];
  const float* dt     = (const float*)d_in[1];
  const float* enc_w  = (const float*)d_in[2];
  const float* enc_b  = (const float*)d_in[3];
  const float* ns_g   = (const float*)d_in[4];
  const float* ns_b   = (const float*)d_in[5];
  const float* cw     = (const float*)d_in[6];
  const float* cb     = (const float*)d_in[7];
  const float* nt_g   = (const float*)d_in[8];
  const float* nt_b   = (const float*)d_in[9];
  const float* E_re   = (const float*)d_in[10];
  const float* E_im   = (const float*)d_in[11];
  const float* Ei_re  = (const float*)d_in[12];
  const float* Ei_im  = (const float*)d_in[13];
  const float* lamf   = (const float*)d_in[14];
  const float* Ws_re  = (const float*)d_in[15];
  const float* Ws_im  = (const float*)d_in[16];
  const float* Wg     = (const float*)d_in[17];
  const float* nu     = (const float*)d_in[18];
  const float* omega  = (const float*)d_in[19];
  const float* w1_re  = (const float*)d_in[20];
  const float* w1_im  = (const float*)d_in[21];
  const float* w2_re  = (const float*)d_in[22];
  const float* w2_im  = (const float*)d_in[23];
  const float* dec_w  = (const float*)d_in[24];
  const float* dec_b  = (const float*)d_in[25];

  // workspace layout (bytes): z | fA(h16) | fB(h16) | g1 | g2 | small
  char* base = (char*)d_ws;
  float*    z  = (float*)base;                               // 12288*768 f32
  _Float16* fA = (_Float16*)(base + 37748736);               // up to 12288*3072 h16
  _Float16* fB = (_Float16*)(base + 37748736 + 75497472);    // up to 9*768*768 h16
  float*    g1 = (float*)(base + 37748736 + 75497472 + 10616832);
  float*    g2 = g1 + (size_t)MTOK * D2;                     // 12288*3072 f32 region
  float*    sm = g2 + (size_t)MTOK * DFF2;                   // xmean|flux|7x small
  float*    xmean = sm;
  float*    flux  = sm + NIMG * D2;
  float*    tsm   = flux + NIMG * D2;                        // 7 * 48*384 floats

  const int TPB = 256;
  auto gcd = [](long long n) { return (unsigned)((n + 255) / 256); };
  auto gblk = [](int M, int N) { return (unsigned)(((M >> 6) * (N >> 5) + 7) / 8); };

  // ---- encoder: unfold -> GEMM -> z (re||im token rows) ----
  unfold_enc<<<gcd((long long)MTOK * KENC), TPB, 0, stream>>>(x, fA);
  cast_f32_h16<<<gcd(D2 * KENC), TPB, 0, stream>>>(enc_w, fB, D2 * KENC); // already [o][k]
  gemm64x32<<<gblk(MTOK, D2), TPB, 0, stream>>>(fA, fB, enc_b, z, MTOK, D2, KENC);

  for (int i = 0; i < 4; ++i) {
    const float* nsg = ns_g + i * D2;   const float* nsb = ns_b + i * D2;
    const float* cwi = cw + (size_t)i * D2 * D2 * 9;
    const float* cbi = cb + i * D2;
    const float* ntg = nt_g + i * D2;   const float* ntb = nt_b + i * D2;

    // spatial branch: LN -> padded NHWC f16 -> implicit-GEMM conv -> residual
    int padN = 48 * 18 * 18 * D2;
    fill_h16<<<gcd(padN), TPB, 0, stream>>>(fA, padN);
    ln_pad<<<MTOK / 8, TPB, 0, stream>>>(z, nsg, nsb, fA);
    build_conv_BT<<<gcd((long long)9 * D2 * D2), TPB, 0, stream>>>(cwi, fB);
    conv3x3_gemm<<<gblk(MTOK, D2), TPB, 0, stream>>>(fA, fB, cbi, g1);
    add_inplace<<<gcd((long long)MTOK * D2), TPB, 0, stream>>>(z, g1, MTOK * D2);

    // x_eig = z @ E (complex as real GEMM)
    build_cplx_BT<<<gcd((long long)4 * DD * DD), TPB, 0, stream>>>(
        E_re + (size_t)i * DD * DD, E_im + (size_t)i * DD * DD, fB, DD, DD);
    cast_f32_h16<<<gcd((long long)MTOK * D2), TPB, 0, stream>>>(z, fA, MTOK * D2);
    gemm64x32<<<gblk(MTOK, D2), TPB, 0, stream>>>(fA, fB, nullptr, g1, MTOK, D2, D2);

    // temporal path (small scalar work)
    mean_hw<<<gcd(NIMG * D2), TPB, 0, stream>>>(g1, xmean);
    flux_scan<<<gcd(BB * DD), TPB, 0, stream>>>(xmean, dt, lamf + i * DD, flux);
    source_gate<<<gcd(NIMG * DD), TPB, 0, stream>>>(
        flux, Ws_re + (size_t)i * DD * DD, Ws_im + (size_t)i * DD * DD,
        Wg + i * DD, nu + i * DD, omega + i * DD, dt, tsm);
    forcing_ut<<<gcd((long long)MTOK * DD), TPB, 0, stream>>>(g1, tsm);
    uout_scan<<<gcd(BB * 256 * DD), TPB, 0, stream>>>(g1, tsm);

    // decoded (df) = u_out @ Ei
    build_cplx_BT<<<gcd((long long)4 * DD * DD), TPB, 0, stream>>>(
        Ei_re + (size_t)i * DD * DD, Ei_im + (size_t)i * DD * DD, fB, DD, DD);
    cast_f32_h16<<<gcd((long long)MTOK * D2), TPB, 0, stream>>>(g1, fA, MTOK * D2);
    gemm64x32<<<gblk(MTOK, D2), TPB, 0, stream>>>(fA, fB, nullptr, g1, MTOK, D2, D2);

    // complex MLP: LN2 -> @W1 -> cgelu -> @W2
    ln_rows<<<MTOK / 8, TPB, 0, stream>>>(g1, ntg, ntb, fA);
    build_cplx_BT<<<gcd((long long)4 * DD * 1536), TPB, 0, stream>>>(
        w1_re + (size_t)i * DD * 1536, w1_im + (size_t)i * DD * 1536, fB, DD, 1536);
    gemm64x32<<<gblk(MTOK, DFF2), TPB, 0, stream>>>(fA, fB, nullptr, g2, MTOK, DFF2, D2);
    gelu_h16<<<gcd((long long)MTOK * DFF2), TPB, 0, stream>>>(g2, fA, MTOK * DFF2);
    build_cplx_BT<<<gcd((long long)4 * 1536 * DD), TPB, 0, stream>>>(
        w2_re + (size_t)i * 1536 * DD, w2_im + (size_t)i * 1536 * DD, fB, 1536, DD);
    gemm64x32<<<gblk(MTOK, D2), TPB, 0, stream>>>(fA, fB, nullptr, g2, MTOK, D2, DFF2);

    // z += df + delta
    residual2<<<gcd((long long)MTOK * D2), TPB, 0, stream>>>(z, g1, g2, MTOK * D2);
  }

  // ---- final decode + pixel shuffle ----
  cast_f32_h16<<<gcd(NDEC * D2), TPB, 0, stream>>>(dec_w, fB, NDEC * D2); // already [o][c]
  cast_f32_h16<<<gcd((long long)MTOK * D2), TPB, 0, stream>>>(z, fA, MTOK * D2);
  gemm64x32<<<gblk(MTOK, NDEC), TPB, 0, stream>>>(fA, fB, dec_b, g2, MTOK, NDEC, D2);
  pixel_shuffle<<<gcd((long long)NIMG * 4 * 256 * 256), TPB, 0, stream>>>(g2, (float*)d_out);
}
